// ArmNet_19902878450319
// MI455X (gfx1250) — matrix operand order repaired and checked by reading, not compiled, not run on hardware
//
#include <hip/hip_runtime.h>
#include <math.h>

// ---------------------------------------------------------------------------
// ArmNet fused pipeline for MI455X (gfx1250, wave32, WMMA)
//   K0 : transform.W f32 -> f16 B panel
//   K0b: remap+pad enc3/dec1/dec2 weights -> f16 WMMA panels (zero-pad K)
//   K1 : encoder: enc1/enc2 VALU, enc3 via WMMA (LDS-staged node rows) -> h f16
//   K2 : transform GEMM [32768x384]@[384x896]: B panel staged to LDS in
//        fragment order via GLOBAL_LOAD_ASYNC_TO_LDS_B128 (ASYNCcnt), ds-fed
//        WMMA, A global + prefetch + one-step pipeline; bias+tanh epilogue
//   K3 : decoder: dec1/dec2 WMMA GEMM pairs, dec3 + limit mapping VALU
// Intermediates (h 25MB, z 59MB, weight panels <1MB) live in d_ws -> L2.
// ---------------------------------------------------------------------------

typedef __attribute__((ext_vector_type(16))) _Float16 v16h;
typedef __attribute__((ext_vector_type(8)))  _Float16 v8h;
typedef __attribute__((ext_vector_type(8)))  float    v8f;

#if defined(__has_builtin)
#if __has_builtin(__builtin_amdgcn_global_load_async_to_lds_b128)
#define USE_ASYNC_LDS 1
#endif
#endif
#ifndef USE_ASYNC_LDS
#define USE_ASYNC_LDS 0
#endif

#if USE_ASYNC_LDS
// exact param type per hipcc diagnostic: 'int __vector(4) __device__ *'
typedef int gv4i __attribute__((vector_size(16)));
typedef __attribute__((address_space(1))) gv4i as1_v4i;   // global source
typedef __attribute__((address_space(3))) gv4i as3_v4i;   // LDS destination
#endif

#define NGRAPH   32768
#define KDIM     384             // 6*64
#define NDIM     896             // 14*64

// workspace layout (bytes, 256B aligned)
#define WT_ELEMS ((size_t)NDIM * KDIM)                    // 344064
#define H_OFF    ((size_t)WT_ELEMS * 2)                   // 688128
#define Z_OFF    (H_OFF + (size_t)NGRAPH * KDIM * 2)      // +25165824
#define DW_OFF   (Z_OFF + (size_t)NGRAPH * NDIM * 2)      // +58720256
// panels (halves): Wu1p 32*96, W2p 32*192, W2q 16*96, Wu2q 16*32,
//                  Wu3p 64*32, W3p 64*96
#define WU1P_OFF (DW_OFF)
#define W2P_OFF  (DW_OFF + 6144)
#define W2Q_OFF  (DW_OFF + 18432)
#define WU2Q_OFF (DW_OFF + 21504)
#define WU3P_OFF (DW_OFF + 22528)
#define W3P_OFF  (DW_OFF + 26624)
// total ws need ~= 84.6 MB

struct EncP {
    const float *W1, *b1, *Wu1, *bu1;   // enc1: lin [16][15], up [16][6]
    const float *W2, *b2, *Wu2, *bu2;   // enc2: lin [32][35], up [32][16]
    const float *W3, *b3, *Wu3, *bu3;   // enc3: lin [64][67], up [64][32]
};
struct DecP {
    const float *W1, *b1, *Wu1, *bu1;   // dec1: lin [32][138], up [32][66]
    const float *W2, *b2, *Wu2, *bu2;   // dec2: lin [16][70],  up [16][32]
    const float *W3, *b3, *Wu3, *bu3;   // dec3: lin [1][38],   up [1][16]
};

__device__ __forceinline__ float lrelu(float s) { return s > 0.f ? s : 0.01f * s; }

__device__ __forceinline__ v16h cat8(v8h lo, v8h hi) {
    return __builtin_shufflevector(lo, hi, 0, 1, 2, 3, 4, 5, 6, 7,
                                           8, 9, 10, 11, 12, 13, 14, 15);
}
__device__ __forceinline__ v8f wmma_f16(v16h a, v16h b, v8f c) {
    return __builtin_amdgcn_wmma_f32_16x16x32_f16(false, a, false, b,
                                                  (short)0, c, false, false);
}

// --------------------------------------------------------------------------
// K0: transform.W f32 [896][384] ([N][K] row-major) -> f16.
// --------------------------------------------------------------------------
__global__ void wconv_kernel(const float* __restrict__ W,
                             _Float16* __restrict__ Wt, int n) {
    int i = blockIdx.x * 256 + threadIdx.x;
    if (i < n) Wt[i] = (_Float16)W[i];
}

// --------------------------------------------------------------------------
// K0b: enc3 + decoder weight remap into zero-padded WMMA panels.
// dec1 node row (96): [z 0..63, lo 64, up 65, et_in 66..71, pad]
// dec2 node row (48): [d1 0..31, et_in 32..37, pad]
// enc3 node row (48): [h2 0..31, ea_in 32..34, pad]
// lin A row = [noderow[j], noderow[i]] contiguous in LDS.
// --------------------------------------------------------------------------
__global__ void prepw_kernel(const float* __restrict__ W1d, const float* __restrict__ Wu1d,
                             const float* __restrict__ W2d, const float* __restrict__ Wu2d,
                             const float* __restrict__ W3e, const float* __restrict__ Wu3e,
                             _Float16* __restrict__ Wu1p, _Float16* __restrict__ W2p,
                             _Float16* __restrict__ W2q,  _Float16* __restrict__ Wu2q,
                             _Float16* __restrict__ Wu3p, _Float16* __restrict__ W3p) {
    int i = blockIdx.x * 256 + threadIdx.x;
    if (i < 3072) {                       // Wu1p [32][96]
        int c = i / 96, k = i - c * 96;
        Wu1p[i] = (_Float16)(k < 66 ? Wu1d[c * 66 + k] : 0.f);
    }
    int j = i - 3072;
    if (j >= 0 && j < 6144) {             // W2p [32][192]
        int c = j / 192, k = j - c * 192;
        float v = 0.f;
        if (k < 64)            v = W1d[c * 138 + 66 + k];        // z_j
        else if (k == 64)      v = W1d[c * 138 + 130];           // lo_j
        else if (k == 65)      v = W1d[c * 138 + 131];           // up_j
        else if (k >= 96 && k < 160) v = W1d[c * 138 + (k - 96)]; // z_i
        else if (k == 160)     v = W1d[c * 138 + 64];            // lo_i
        else if (k == 161)     v = W1d[c * 138 + 65];            // up_i
        else if (k >= 162 && k < 168) v = W1d[c * 138 + 132 + (k - 162)]; // et
        W2p[j] = (_Float16)v;
    }
    int q = i - 9216;
    if (q >= 0 && q < 1536) {             // W2q [16][96]
        int c = q / 96, k = q - c * 96;
        float v = 0.f;
        if (k < 32)                  v = W2d[c * 70 + 32 + k];        // d1_j
        else if (k >= 48 && k < 80)  v = W2d[c * 70 + (k - 48)];      // d1_i
        else if (k >= 80 && k < 86)  v = W2d[c * 70 + 64 + (k - 80)]; // et
        W2q[q] = (_Float16)v;
    }
    int u = i - 10752;
    if (u >= 0 && u < 512) Wu2q[u] = (_Float16)Wu2d[u];  // Wu2q [16][32]
    int w3 = i - 11264;
    if (w3 >= 0 && w3 < 2048) Wu3p[w3] = (_Float16)Wu3e[w3]; // Wu3p [64][32]
    int w4 = i - 13312;
    if (w4 >= 0 && w4 < 6144) {           // W3p [64][96]
        int c = w4 / 96, k = w4 - c * 96;
        float v = 0.f;
        if (k < 32)                  v = W3e[c * 67 + 32 + k];        // h2_j
        else if (k >= 48 && k < 80)  v = W3e[c * 67 + (k - 48)];      // h2_i
        else if (k >= 80 && k < 83)  v = W3e[c * 67 + 64 + (k - 80)]; // ea
        W3p[w4] = (_Float16)v;
    }
}

// --------------------------------------------------------------------------
// K1: encoder. 8 graphs/WG. enc1/enc2 wave-per-graph VALU; enc3 WMMA.
// --------------------------------------------------------------------------
__global__ __launch_bounds__(256) void encoder_kernel(
    const float* __restrict__ x, const float* __restrict__ eas,
    EncP p, const _Float16* __restrict__ Wu3p, const _Float16* __restrict__ W3p,
    _Float16* __restrict__ Hb) {
    __shared__ float sm[8 * 344];
    __shared__ __align__(16) _Float16 NR3[48 * 48];   // enc3 node rows
    __shared__ __align__(16) _Float16 UP3[48 * 64];   // up staging
    __shared__ __align__(16) _Float16 LIN3[48 * 64];  // lin staging
    const int tid  = threadIdx.x;
    const int lane = tid & 31;
    const int wid  = tid >> 5;
    const int mr   = lane & 15;
    const int hi   = lane >> 4;
    const int g0   = blockIdx.x * 8;
    const int g    = g0 + wid;
    float* base = sm + wid * 344;
    float* xg = base;          // 36
    float* ea = base + 36;     // 15
    float* h1 = base + 52;     // 96
    float* h2 = base + 148;    // 192

    {
        int4 z4 = {0, 0, 0, 0};
        for (int i = tid; i < 288; i += 256) ((int4*)NR3)[i] = z4;
    }
    for (int i = lane; i < 36; i += 32) xg[i] = x[(size_t)g * 36 + i];
    for (int i = lane; i < 15; i += 32) ea[i] = eas[(size_t)g * 15 + i];
    __syncthreads();

    // enc1: 16 channels, two nodes per pass
    {
        const int c = mr, po = hi;
        for (int np = 0; np < 3; ++np) {
            const int n = 2 * np + po;
            float up = p.bu1[c];
#pragma unroll
            for (int k = 0; k < 6; ++k) up += p.Wu1[c * 6 + k] * xg[n * 6 + k];
            float m = 0.f;
            if (n > 0) {
                float s = p.b1[c];
#pragma unroll
                for (int k = 0; k < 6; ++k) s += p.W1[c * 15 + k] * xg[n * 6 + k];
#pragma unroll
                for (int k = 0; k < 6; ++k) s += p.W1[c * 15 + 6 + k] * xg[(n - 1) * 6 + k];
#pragma unroll
                for (int k = 0; k < 3; ++k) s += p.W1[c * 15 + 12 + k] * ea[(n - 1) * 3 + k];
                m = lrelu(s);
            }
            h1[n * 16 + c] = up + m;
        }
    }
    __syncthreads();

    // enc2: 32 channels
    {
        const int c = lane;
        for (int n = 0; n < 6; ++n) {
            float up = p.bu2[c];
#pragma unroll
            for (int k = 0; k < 16; ++k) up += p.Wu2[c * 16 + k] * h1[n * 16 + k];
            float m = 0.f;
            if (n > 0) {
                float s = p.b2[c];
#pragma unroll
                for (int k = 0; k < 16; ++k) s += p.W2[c * 35 + k] * h1[n * 16 + k];
#pragma unroll
                for (int k = 0; k < 16; ++k) s += p.W2[c * 35 + 16 + k] * h1[(n - 1) * 16 + k];
#pragma unroll
                for (int k = 0; k < 3; ++k) s += p.W2[c * 35 + 32 + k] * ea[(n - 1) * 3 + k];
                m = lrelu(s);
            }
            h2[n * 32 + c] = up + m;
        }
    }
    __syncthreads();

    // stage enc3 node rows: [h2(32), ea_in(3), pad] stride 48
    for (int i = lane; i < 192; i += 32)
        NR3[(wid * 6 + (i >> 5)) * 48 + (i & 31)] = (_Float16)h2[i];
    for (int i = lane; i < 15; i += 32) {
        int ne = i / 3, c = i - ne * 3;
        NR3[(wid * 6 + ne + 1) * 48 + 32 + c] = (_Float16)ea[i];
    }
    __syncthreads();

    // enc3 up GEMM: [48][32]@[32][64]
    for (int job = wid; job < 12; job += 8) {
        const int mt = job >> 2, nt = job & 3;
        const int r = mt * 16 + mr;
        const _Float16* ab = NR3 + r * 48 + hi * 16;
        const _Float16* bb = Wu3p + (nt * 16 + mr) * 32 + hi * 16;
        v8f acc = wmma_f16(cat8(*(const v8h*)(ab), *(const v8h*)(ab + 8)),
                           cat8(*(const v8h*)(bb), *(const v8h*)(bb + 8)),
                           (v8f){});
#pragma unroll
        for (int e = 0; e < 8; ++e)
            UP3[(mt * 16 + hi * 8 + e) * 64 + nt * 16 + mr] = (_Float16)acc[e];
    }
    // enc3 lin GEMM: [48][96]@[96][64] over edge rows
    for (int job = wid; job < 12; job += 8) {
        const int mt = job >> 2, nt = job & 3;
        const int r = mt * 16 + mr;          // edge row
        const int gg = r / 5;
        const _Float16* ab = NR3 + (r + gg) * 48 + hi * 16;
        const _Float16* bb = W3p + (nt * 16 + mr) * 96 + hi * 16;
        v8f acc = {};
#pragma unroll
        for (int kt = 0; kt < 3; ++kt)
            acc = wmma_f16(cat8(*(const v8h*)(ab + kt * 32), *(const v8h*)(ab + kt * 32 + 8)),
                           cat8(*(const v8h*)(bb + kt * 32), *(const v8h*)(bb + kt * 32 + 8)),
                           acc);
#pragma unroll
        for (int e = 0; e < 8; ++e)
            LIN3[(mt * 16 + hi * 8 + e) * 64 + nt * 16 + mr] = (_Float16)acc[e];
    }
    __syncthreads();

    // combine -> Hb
    for (int idx = tid; idx < 48 * 64; idx += 256) {
        const int rr = idx >> 6, c = idx & 63;
        const int gg = rr / 6, n = rr - gg * 6;
        float v = p.bu3[c] + (float)UP3[idx];
        if (n > 0) v += lrelu((float)LIN3[(rr - gg - 1) * 64 + c] + p.b3[c]);
        Hb[(size_t)(g0 + gg) * KDIM + n * 64 + c] = (_Float16)v;
    }
}

// --------------------------------------------------------------------------
// K2: transform GEMM + bias + tanh. 64x64 tile per WG. B panel staged to
// LDS in fragment order: chunk[(ntile*12+kt)*32+lane] = the 16 halves that
// lane consumes -> WMMAs fed by ds_load_b128. Staging uses async global->LDS
// (ASYNCcnt) when the toolchain exposes it; VGPR copy otherwise.
// --------------------------------------------------------------------------
__global__ __launch_bounds__(256) void transform_kernel(
    const _Float16* __restrict__ Hb, const _Float16* __restrict__ Wt,
    const float* __restrict__ bias, _Float16* __restrict__ Zb) {
    __shared__ __align__(16) _Float16 BL[4 * 12 * 32 * 16];   // 49152 B
    const int tid  = threadIdx.x;
    const int lane = tid & 31;
    const int wid  = tid >> 5;
    const int mr   = lane & 15;
    const int hi   = lane >> 4;

    const int mBase = blockIdx.x * 64 + (wid & 3) * 16;
    const int nBase = blockIdx.y * 64;

    // warm the A stream while B is being staged (global_prefetch_b8)
    const _Float16* arow = Hb + (size_t)(mBase + mr) * KDIM + hi * 16;
    __builtin_prefetch(arow, 0, 1);
    __builtin_prefetch(arow + 128, 0, 1);
    __builtin_prefetch(arow + 256, 0, 1);

    // stage B panel swizzled (1536 chunks x 32B)
    for (int c = tid; c < 1536; c += 256) {
        const int ntl = c / 384;
        const int rem = c - ntl * 384;
        const int kt  = rem >> 5;
        const int ln  = rem & 31;
        const int chi = ln >> 4, cmr = ln & 15;
        const _Float16* src = Wt + (size_t)(nBase + ntl * 16 + cmr) * KDIM + kt * 32 + chi * 16;
        _Float16* dst = BL + (size_t)c * 16;
#if USE_ASYNC_LDS
        // per-lane async copy: LDS[dst+off] = MEM[src+off], 2 x b128 per chunk
        __builtin_amdgcn_global_load_async_to_lds_b128(
            (as1_v4i*)(size_t)src, (as3_v4i*)(unsigned)(size_t)dst, 0, 0);
        __builtin_amdgcn_global_load_async_to_lds_b128(
            (as1_v4i*)(size_t)src, (as3_v4i*)(unsigned)(size_t)dst, 16, 0);
#else
        *(v8h*)dst       = *(const v8h*)src;
        *(v8h*)(dst + 8) = *(const v8h*)(src + 8);
#endif
    }
#if USE_ASYNC_LDS
#if defined(__has_builtin) && __has_builtin(__builtin_amdgcn_s_wait_asynccnt)
    __builtin_amdgcn_s_wait_asynccnt(0);
#else
    asm volatile("s_wait_asynccnt 0x0" ::: "memory");
#endif
#endif
    __syncthreads();

    const int t0 = (wid >> 2) * 2;     // this wave's ntiles {t0, t0+1}

    v8f acc[2] = {};
    v8h a0 = *(const v8h*)(arow);
    v8h a1 = *(const v8h*)(arow + 8);
#pragma unroll
    for (int kt = 0; kt < 12; ++kt) {
        v8h na0, na1;
        if (kt < 11) {
            na0 = *(const v8h*)(arow + (kt + 1) * 32);
            na1 = *(const v8h*)(arow + (kt + 1) * 32 + 8);
        }
        v16h aF = cat8(a0, a1);
#pragma unroll
        for (int q = 0; q < 2; ++q) {
            const _Float16* bp = BL + (size_t)(((t0 + q) * 12 + kt) * 32 + lane) * 16;
            acc[q] = wmma_f16(aF, cat8(*(const v8h*)bp, *(const v8h*)(bp + 8)), acc[q]);
        }
        if (kt < 11) { a0 = na0; a1 = na1; }
    }

#pragma unroll
    for (int q = 0; q < 2; ++q) {
        const int col = nBase + (t0 + q) * 16 + mr;
        const float bv = bias[col];
#pragma unroll
        for (int e = 0; e < 8; ++e) {
            const int row = mBase + hi * 8 + e;
            Zb[(size_t)row * NDIM + col] = (_Float16)tanhf(acc[q][e] + bv);
        }
    }
}

// --------------------------------------------------------------------------
// K3: decoder, 8 graphs/WG. dec1/dec2 WMMA, dec3 VALU.
// --------------------------------------------------------------------------
__global__ __launch_bounds__(256) void decoder_kernel(
    const _Float16* __restrict__ Zb, const float* __restrict__ lower,
    const float* __restrict__ upper, const float* __restrict__ eat,
    DecP p,
    const _Float16* __restrict__ Wu1p, const _Float16* __restrict__ W2p,
    const _Float16* __restrict__ W2q,  const _Float16* __restrict__ Wu2q,
    float* __restrict__ out) {
    __shared__ __align__(16) unsigned char smem[53760];
    _Float16* NR   = (_Float16*)(smem);
    _Float16* NR2  = (_Float16*)(smem + 21504);
    _Float16* UP1  = (_Float16*)(smem + 32256);
    _Float16* LIN1 = (_Float16*)(smem + 39424);
    float*    D2S  = (float*)   (smem + 46592);

    const int tid  = threadIdx.x;
    const int lane = tid & 31;
    const int wid  = tid >> 5;
    const int mr   = lane & 15;
    const int hi   = lane >> 4;
    const int g0   = blockIdx.x * 8;

    {
        int4 z4 = {0, 0, 0, 0};
        for (int i = tid; i < 2016; i += 256) ((int4*)smem)[i] = z4;
    }
    __syncthreads();

    {
        const int g = g0 + wid;
        const _Float16* zrow = Zb + (size_t)g * NDIM;
        for (int i = lane; i < 896; i += 32)
            NR[(wid * 14 + (i >> 6)) * 96 + (i & 63)] = zrow[i];
        if (lane < 14) {
            NR[(wid * 14 + lane) * 96 + 64] = (_Float16)lower[(size_t)g * 14 + lane];
            NR[(wid * 14 + lane) * 96 + 65] = (_Float16)upper[(size_t)g * 14 + lane];
        }
        for (int i = lane; i < 78; i += 32) {
            int ne = i / 6, c = i - ne * 6;
            NR[(wid * 14 + ne + 1) * 96 + 66 + c] = (_Float16)eat[(size_t)g * 78 + i];
        }
    }
    __syncthreads();

    for (int job = wid; job < 14; job += 8) {          // dec1 up
        const int mt = job >> 1, nt = job & 1;
        const int r = mt * 16 + mr;
        const _Float16* ab = NR + r * 96 + hi * 16;
        const _Float16* bb = Wu1p + (nt * 16 + mr) * 96 + hi * 16;
        v8f acc = {};
#pragma unroll
        for (int kt = 0; kt < 3; ++kt)
            acc = wmma_f16(cat8(*(const v8h*)(ab + kt * 32), *(const v8h*)(ab + kt * 32 + 8)),
                           cat8(*(const v8h*)(bb + kt * 32), *(const v8h*)(bb + kt * 32 + 8)),
                           acc);
#pragma unroll
        for (int e = 0; e < 8; ++e)
            UP1[(mt * 16 + hi * 8 + e) * 32 + nt * 16 + mr] = (_Float16)acc[e];
    }
    for (int job = wid; job < 14; job += 8) {          // dec1 lin
        const int mt = job >> 1, nt = job & 1;
        const int r = mt * 16 + mr;
        const int gg = r / 13;
        const _Float16* ab = NR + (r + gg) * 96 + hi * 16;
        const _Float16* bb = W2p + (nt * 16 + mr) * 192 + hi * 16;
        v8f acc = {};
#pragma unroll
        for (int kt = 0; kt < 6; ++kt)
            acc = wmma_f16(cat8(*(const v8h*)(ab + kt * 32), *(const v8h*)(ab + kt * 32 + 8)),
                           cat8(*(const v8h*)(bb + kt * 32), *(const v8h*)(bb + kt * 32 + 8)),
                           acc);
#pragma unroll
        for (int e = 0; e < 8; ++e)
            LIN1[(mt * 16 + hi * 8 + e) * 32 + nt * 16 + mr] = (_Float16)acc[e];
    }
    __syncthreads();

    for (int idx = tid; idx < 112 * 32; idx += 256) {  // combine dec1 -> NR2
        const int r = idx >> 5, c = idx & 31;
        const int gg = r / 14, n = r - gg * 14;
        float v = p.bu1[c] + (float)UP1[idx];
        if (n > 0) v += lrelu((float)LIN1[(r - gg - 1) * 32 + c] + p.b1[c]);
        NR2[r * 48 + c] = (_Float16)v;
    }
    for (int idx = tid; idx < 104 * 6; idx += 256) {
        const int e = idx / 6, c = idx - e * 6;
        const int gg = e / 13, n = e - gg * 13 + 1;
        const int r = gg * 14 + n;
        NR2[r * 48 + 32 + c] = NR[r * 96 + 66 + c];
    }
    __syncthreads();

    for (int job = wid; job < 7; job += 8) {           // dec2 up
        const int mt = job;
        const int r = mt * 16 + mr;
        const _Float16* ab = NR2 + r * 48 + hi * 16;
        const _Float16* bb = Wu2q + mr * 32 + hi * 16;
        v8f acc = wmma_f16(cat8(*(const v8h*)(ab), *(const v8h*)(ab + 8)),
                           cat8(*(const v8h*)(bb), *(const v8h*)(bb + 8)),
                           (v8f){});
#pragma unroll
        for (int e = 0; e < 8; ++e)
            UP1[(mt * 16 + hi * 8 + e) * 16 + mr] = (_Float16)acc[e];
    }
    for (int job = wid; job < 7; job += 8) {           // dec2 lin
        const int mt = job;
        const int r = mt * 16 + mr;
        const int gg = r / 13;
        const _Float16* ab = NR2 + (r + gg) * 48 + hi * 16;
        const _Float16* bb = W2q + mr * 96 + hi * 16;
        v8f acc = {};
#pragma unroll
        for (int kt = 0; kt < 3; ++kt)
            acc = wmma_f16(cat8(*(const v8h*)(ab + kt * 32), *(const v8h*)(ab + kt * 32 + 8)),
                           cat8(*(const v8h*)(bb + kt * 32), *(const v8h*)(bb + kt * 32 + 8)),
                           acc);
#pragma unroll
        for (int e = 0; e < 8; ++e)
            LIN1[(mt * 16 + hi * 8 + e) * 16 + mr] = (_Float16)acc[e];
    }
    __syncthreads();

    for (int idx = tid; idx < 112 * 16; idx += 256) {  // combine dec2 -> D2S
        const int r = idx >> 4, c = idx & 15;
        const int gg = r / 14, n = r - gg * 14;
        float v = p.bu2[c] + (float)UP1[idx];
        if (n > 0) v += lrelu((float)LIN1[(r - gg - 1) * 16 + c] + p.b2[c]);
        D2S[idx] = v;
    }
    __syncthreads();

    if (tid < 112) {                                   // dec3 + mapping
        const int r = tid;
        const int gg = r / 14, n = r - gg * 14;
        const float* d2r = D2S + r * 16;
        float up = p.bu3[0];
#pragma unroll
        for (int k = 0; k < 16; ++k) up += p.Wu3[k] * d2r[k];
        float m = 0.f;
        if (n > 0) {
            const float* d2p = D2S + (r - 1) * 16;
            float s = p.b3[0];
#pragma unroll
            for (int k = 0; k < 16; ++k) s += p.W3[k] * d2r[k];
#pragma unroll
            for (int k = 0; k < 16; ++k) s += p.W3[16 + k] * d2p[k];
#pragma unroll
            for (int k = 0; k < 6; ++k) s += p.W3[32 + k] * (float)NR[r * 96 + 66 + k];
            m = lrelu(s);
        }
        const float d = tanhf(up + m);
        const int g = g0 + gg;
        const float lo = lower[(size_t)g * 14 + n];
        const float hb = upper[(size_t)g * 14 + n];
        out[(size_t)g * 14 + n] = lo + (hb - lo) * (d + 1.0f) * 0.5f;
    }
}

// --------------------------------------------------------------------------
// Host launcher
// --------------------------------------------------------------------------
extern "C" void kernel_launch(void* const* d_in, const int* in_sizes, int n_in,
                              void* d_out, int out_size, void* d_ws, size_t ws_size,
                              hipStream_t stream) {
    const float* x      = (const float*)d_in[0];
    const float* ea_src = (const float*)d_in[2];
    const float* ea_tgt = (const float*)d_in[4];
    const float* lower  = (const float*)d_in[5];
    const float* upper  = (const float*)d_in[6];

    const float* P[26];
    for (int i = 0; i < 26; ++i) P[i] = (const float*)d_in[7 + i];

    EncP ep;
    DecP dp;
    const float *tW, *tb;
    if (in_sizes[7] == 240) {
        // dict-insertion order: enc1, enc2, enc3, transform, dec1, dec2, dec3
        ep = { P[0],  P[1],  P[2],  P[3],
               P[4],  P[5],  P[6],  P[7],
               P[8],  P[9],  P[10], P[11] };
        tW = P[12]; tb = P[13];
        dp = { P[14], P[15], P[16], P[17],
               P[18], P[19], P[20], P[21],
               P[22], P[23], P[24], P[25] };
    } else {
        // pytree-sorted order: dec1, dec2, dec3, enc1, enc2, enc3, transform
        dp = { P[0],  P[1],  P[2],  P[3],
               P[4],  P[5],  P[6],  P[7],
               P[8],  P[9],  P[10], P[11] };
        ep = { P[12], P[13], P[14], P[15],
               P[16], P[17], P[18], P[19],
               P[20], P[21], P[22], P[23] };
        tW = P[24]; tb = P[25];
    }

    _Float16* Wt   = (_Float16*)((char*)d_ws);
    _Float16* Hb   = (_Float16*)((char*)d_ws + H_OFF);
    _Float16* Zb   = (_Float16*)((char*)d_ws + Z_OFF);
    _Float16* Wu1p = (_Float16*)((char*)d_ws + WU1P_OFF);
    _Float16* W2p  = (_Float16*)((char*)d_ws + W2P_OFF);
    _Float16* W2q  = (_Float16*)((char*)d_ws + W2Q_OFF);
    _Float16* Wu2q = (_Float16*)((char*)d_ws + WU2Q_OFF);
    _Float16* Wu3p = (_Float16*)((char*)d_ws + WU3P_OFF);
    _Float16* W3p  = (_Float16*)((char*)d_ws + W3P_OFF);

    wconv_kernel<<<dim3((int)((WT_ELEMS + 255) / 256)), dim3(256), 0, stream>>>(
        tW, Wt, (int)WT_ELEMS);
    prepw_kernel<<<dim3(76), dim3(256), 0, stream>>>(
        dp.W1, dp.Wu1, dp.W2, dp.Wu2, ep.W3, ep.Wu3,
        Wu1p, W2p, W2q, Wu2q, Wu3p, W3p);
    encoder_kernel<<<dim3(NGRAPH / 8), dim3(256), 0, stream>>>(
        x, ea_src, ep, Wu3p, W3p, Hb);
    transform_kernel<<<dim3(NGRAPH / 64, NDIM / 64), dim3(256), 0, stream>>>(
        Hb, Wt, tb, Zb);
    decoder_kernel<<<dim3(NGRAPH / 8), dim3(256), 0, stream>>>(
        Zb, lower, upper, ea_tgt, dp, Wu1p, W2p, W2q, Wu2q, (float*)d_out);
}
